// AttentionDecoder_44719199486044
// MI455X (gfx1250) — matrix operand order repaired and checked by reading, not compiled, not run on hardware
//
#include <hip/hip_runtime.h>
#include <hip/hip_fp16.h>

typedef __attribute__((ext_vector_type(16))) _Float16 v16h;
typedef __attribute__((ext_vector_type(8)))  float    v8f;

#define B_   64
#define T_   512
#define E_   512
#define H_   512
#define A_   256
#define V_   8000
#define EMB_ 64
#define L_   64
#define X_   (EMB_ + E_)   // 576
#define G_   (3 * H_)      // 1536

// ---------------------------------------------------------------- utilities

__global__ __launch_bounds__(256) void cvt_f32_to_f16(const float* __restrict__ s,
                                                      _Float16* __restrict__ d, int n) {
  int i = blockIdx.x * blockDim.x + threadIdx.x;
  if (i < n) d[i] = (_Float16)s[i];
}

__global__ __launch_bounds__(256) void zero_f32(float* __restrict__ p, int n) {
  int i = blockIdx.x * blockDim.x + threadIdx.x;
  if (i < n) p[i] = 0.0f;
}

__global__ __launch_bounds__(256) void zero_f16(_Float16* __restrict__ p, int n) {
  int i = blockIdx.x * blockDim.x + threadIdx.x;
  if (i < n) p[i] = (_Float16)0.0f;
}

// embs[b,l,:] = emb_table[prev_char], prev = (l==0) ? SOS(1) : targets[b,l-1]
__global__ __launch_bounds__(256) void gather_embs(const int* __restrict__ targets,
                                                   const float* __restrict__ emb_table,
                                                   float* __restrict__ embs) {
  int i  = blockIdx.x * blockDim.x + threadIdx.x;   // over B*L*EMB
  int e  = i & (EMB_ - 1);
  int bl = i >> 6;
  int l  = bl & (L_ - 1);
  int b  = bl >> 6;
  int prev = (l == 0) ? 1 : targets[b * L_ + l - 1];
  embs[i] = emb_table[prev * EMB_ + e];
}

// ---------------------------------------------------- fragment load helpers

__device__ __forceinline__ v16h load_a_frag_f16(const _Float16* __restrict__ arow, int k) {
  // A 16x32 f16 layout: lanes<16 hold K {k..k+7, k+16..k+23} of their row;
  // lanes>=16 hold K {k+8..k+15, k+24..k+31} (arow pre-offset by hi?8:0).
  v16h a;
#pragma unroll
  for (int j = 0; j < 8; ++j) {
    a[j]     = arow[k + j];
    a[j + 8] = arow[k + 16 + j];
  }
  return a;
}

__device__ __forceinline__ v16h load_b_frag_f16(const _Float16* __restrict__ wrow, int k) {
  // B 32x16 layout: lanes<16 hold K {k..k+7, k+16..k+23} of column N=lane;
  // lanes>=16 hold K {k+8..k+15, k+24..k+31} (wrow pre-offset by hi?8:0).
  v16h b;
#pragma unroll
  for (int j = 0; j < 8; ++j) {
    b[j]     = wrow[k + j];
    b[j + 8] = wrow[k + 16 + j];
  }
  return b;
}

// ------------------------------------------------ generic WMMA GEMM (f32 A)
// C(MxN) = A(MxK,f32) * W(NxK,f16)^T ; one-time proj_enc GEMM only.
// grid.x = N/64 (4 waves x 16 cols), grid.y = M/16, block = 128.
__global__ __launch_bounds__(128) void gemm_wmma(const float* __restrict__ Af, int lda,
                                                 const _Float16* __restrict__ Wh, int ldw,
                                                 float* __restrict__ C, long long ldc, int K) {
  const int wave = threadIdx.x >> 5;
  const int lane = threadIdx.x & 31;
  const int r    = lane & 15;
  const bool hi  = lane >= 16;
  const int m0   = blockIdx.y << 4;
  const int n0   = (blockIdx.x << 6) + (wave << 4);

  v8f acc = {};
  const float*    arow = Af + (size_t)(m0 + r) * lda + (hi ? 8 : 0);
  const _Float16* wrow = Wh + (size_t)(n0 + r) * ldw + (hi ? 8 : 0);

  for (int k = 0; k < K; k += 32) {
    v16h a;
#pragma unroll
    for (int j = 0; j < 8; ++j) {
      a[j]     = (_Float16)arow[k + j];
      a[j + 8] = (_Float16)arow[k + 16 + j];
    }
    v16h b = load_b_frag_f16(wrow, k);
    acc = __builtin_amdgcn_wmma_f32_16x16x32_f16(false, a, false, b,
                                                 (short)0, acc, false, false);
  }

#pragma unroll
  for (int vr = 0; vr < 8; ++vr) {
    int m = m0 + vr + (hi ? 8 : 0);
    C[(size_t)m * (size_t)ldc + n0 + r] = acc[vr];
  }
}

// ------------------------------------ skinny WMMA GEMM: M == 64, f16 A
// C(64xN) = Ah(64xK,f16) * W(NxK,f16)^T [+ bias]. Each wave owns one n-tile
// and all 4 m-tiles: per K-step, 1 B-fragment load feeds 4 WMMAs, so W is
// streamed exactly once per GEMM. grid.x = N/64, block = 128 (4 waves).
__global__ __launch_bounds__(128) void gemm_skinny(const _Float16* __restrict__ Ah, int lda,
                                                   const _Float16* __restrict__ Wh, int ldw,
                                                   const float* __restrict__ bias,
                                                   float* __restrict__ C, long long ldc, int K) {
  const int wave = threadIdx.x >> 5;
  const int lane = threadIdx.x & 31;
  const int r    = lane & 15;
  const bool hi  = lane >= 16;
  const int n0   = (blockIdx.x << 6) + (wave << 4);

  v8f acc0 = {}, acc1 = {}, acc2 = {}, acc3 = {};
  const int aoff = hi ? 8 : 0;
  const _Float16* arow0 = Ah + (size_t)(r)      * lda + aoff;
  const _Float16* arow1 = Ah + (size_t)(16 + r) * lda + aoff;
  const _Float16* arow2 = Ah + (size_t)(32 + r) * lda + aoff;
  const _Float16* arow3 = Ah + (size_t)(48 + r) * lda + aoff;
  const _Float16* wrow  = Wh + (size_t)(n0 + r) * ldw + (hi ? 8 : 0);

  for (int k = 0; k < K; k += 32) {
    v16h b  = load_b_frag_f16(wrow, k);
    v16h a0 = load_a_frag_f16(arow0, k);
    v16h a1 = load_a_frag_f16(arow1, k);
    v16h a2 = load_a_frag_f16(arow2, k);
    v16h a3 = load_a_frag_f16(arow3, k);
    acc0 = __builtin_amdgcn_wmma_f32_16x16x32_f16(false, a0, false, b, (short)0, acc0, false, false);
    acc1 = __builtin_amdgcn_wmma_f32_16x16x32_f16(false, a1, false, b, (short)0, acc1, false, false);
    acc2 = __builtin_amdgcn_wmma_f32_16x16x32_f16(false, a2, false, b, (short)0, acc2, false, false);
    acc3 = __builtin_amdgcn_wmma_f32_16x16x32_f16(false, a3, false, b, (short)0, acc3, false, false);
  }

  const float bv = bias ? bias[n0 + r] : 0.0f;
  const int   mb = hi ? 8 : 0;
#pragma unroll
  for (int vr = 0; vr < 8; ++vr) {
    int nidx = n0 + r;
    C[(size_t)(vr + mb)      * (size_t)ldc + nidx] = acc0[vr] + bv;
    C[(size_t)(16 + vr + mb) * (size_t)ldc + nidx] = acc1[vr] + bv;
    C[(size_t)(32 + vr + mb) * (size_t)ldc + nidx] = acc2[vr] + bv;
    C[(size_t)(48 + vr + mb) * (size_t)ldc + nidx] = acc3[vr] + bv;
  }
}

// -------------------------------------------- fused attention per batch b
__global__ __launch_bounds__(256) void attn_fused(const float* __restrict__ proj_enc,
                                                  const float* __restrict__ proj_dec,
                                                  const float* __restrict__ vvec,
                                                  const float* __restrict__ enc_out,
                                                  const float* __restrict__ embs,
                                                  float* __restrict__ x,
                                                  float* __restrict__ attn_out, int l) {
  __shared__ float s_pd[A_];
  __shared__ float s_v[A_];
  __shared__ float s_e[T_];
  __shared__ float s_red[256];
  const int tid  = threadIdx.x;
  const int b    = blockIdx.x;
  const int wave = tid >> 5, lane = tid & 31;

  s_pd[tid] = proj_dec[b * A_ + tid];
  s_v[tid]  = vvec[tid];
  __syncthreads();

  // energy: each wave covers 64 t's; lanes split the A=256 dot, shfl-reduce
  for (int tw = 0; tw < T_ / 8; ++tw) {
    int t = wave * (T_ / 8) + tw;
    const float* pe = proj_enc + ((size_t)b * T_ + t) * A_;
    float sum = 0.0f;
#pragma unroll
    for (int a = 0; a < A_ / 32; ++a) {
      int ai = lane + a * 32;
      sum += s_v[ai] * tanhf(pe[ai] + s_pd[ai]);
    }
#pragma unroll
    for (int off = 16; off > 0; off >>= 1) sum += __shfl_xor(sum, off, 32);
    if (lane == 0) s_e[t] = sum;
  }
  __syncthreads();

  // softmax over T=512 (2 elements per thread)
  s_red[tid] = fmaxf(s_e[tid], s_e[tid + 256]);
  __syncthreads();
  for (int s = 128; s > 0; s >>= 1) {
    if (tid < s) s_red[tid] = fmaxf(s_red[tid], s_red[tid + s]);
    __syncthreads();
  }
  float gmax = s_red[0];
  __syncthreads();
  float p0 = expf(s_e[tid] - gmax), p1 = expf(s_e[tid + 256] - gmax);
  s_red[tid] = p0 + p1;
  __syncthreads();
  for (int s = 128; s > 0; s >>= 1) {
    if (tid < s) s_red[tid] += s_red[tid + s];
    __syncthreads();
  }
  float inv = 1.0f / s_red[0];
  __syncthreads();
  p0 *= inv; p1 *= inv;
  s_e[tid] = p0; s_e[tid + 256] = p1;
  size_t abase = ((size_t)b * L_ + l) * T_;
  attn_out[abase + tid]       = p0;
  attn_out[abase + 256 + tid] = p1;
  __syncthreads();

  // context (coalesced across e; enc_out rows stay in L2)
  for (int e = tid; e < E_; e += 256) {
    float acc = 0.0f;
    const float* eo = enc_out + (size_t)b * T_ * E_ + e;
#pragma unroll 4
    for (int t = 0; t < T_; ++t) acc += s_e[t] * eo[(size_t)t * E_];
    x[b * X_ + EMB_ + e] = acc;
  }
  if (tid < EMB_) x[b * X_ + tid] = embs[((size_t)b * L_ + l) * EMB_ + tid];
}

// --------------------------------------- GRU update; also emits f16 hidden
__global__ __launch_bounds__(256) void gru_update(const float* __restrict__ gi,
                                                  const float* __restrict__ gh,
                                                  float* __restrict__ h,
                                                  _Float16* __restrict__ h16) {
  int i = blockIdx.x * blockDim.x + threadIdx.x;  // over B*H
  int b = i >> 9, hh = i & (H_ - 1);
  int base = b * G_;
  float rr = 1.0f / (1.0f + expf(-(gi[base + hh] + gh[base + hh])));
  float zz = 1.0f / (1.0f + expf(-(gi[base + H_ + hh] + gh[base + H_ + hh])));
  float nn = tanhf(gi[base + 2 * H_ + hh] + rr * gh[base + 2 * H_ + hh]);
  float hn = (1.0f - zz) * nn + zz * h[i];
  h[i]   = hn;
  h16[i] = (_Float16)hn;
}

// --------------------------------------------------------------- launcher

extern "C" void kernel_launch(void* const* d_in, const int* in_sizes, int n_in,
                              void* d_out, int out_size, void* d_ws, size_t ws_size,
                              hipStream_t stream) {
  (void)in_sizes; (void)n_in; (void)out_size; (void)ws_size;

  const float* enc_out   = (const float*)d_in[0];
  const int*   targets   = (const int*)d_in[1];
  const float* W_enc     = (const float*)d_in[2];
  const float* W_dec     = (const float*)d_in[3];
  const float* v         = (const float*)d_in[4];
  const float* emb_table = (const float*)d_in[5];
  const float* W_ih      = (const float*)d_in[6];
  const float* W_hh      = (const float*)d_in[7];
  const float* b_ih      = (const float*)d_in[8];
  const float* b_hh      = (const float*)d_in[9];
  const float* W_out     = (const float*)d_in[10];
  const float* b_out     = (const float*)d_in[11];

  float* out_logits = (float*)d_out;                           // (B,L,V)
  float* out_attns  = (float*)d_out + (size_t)B_ * L_ * V_;    // (B,L,T)

  char* ws = (char*)d_ws;
  size_t off = 0;
  auto alloc = [&](size_t bytes) {
    size_t o = off;
    off += (bytes + 255) & ~(size_t)255;
    return o;
  };
  float*    proj_enc = (float*)   (ws + alloc((size_t)B_ * T_ * A_ * 4));   // 33.5 MB (L2-resident)
  _Float16* W_enc_h  = (_Float16*)(ws + alloc((size_t)A_ * E_ * 2));
  _Float16* W_dec_h  = (_Float16*)(ws + alloc((size_t)A_ * H_ * 2));
  _Float16* W_ih_h   = (_Float16*)(ws + alloc((size_t)G_ * X_ * 2));
  _Float16* W_hh_h   = (_Float16*)(ws + alloc((size_t)G_ * H_ * 2));
  _Float16* W_out_h  = (_Float16*)(ws + alloc((size_t)V_ * H_ * 2));        // 8 MB
  float*    embs = (float*)   (ws + alloc((size_t)B_ * L_ * EMB_ * 4));
  float*    h    = (float*)   (ws + alloc((size_t)B_ * H_ * 4));
  _Float16* h16  = (_Float16*)(ws + alloc((size_t)B_ * H_ * 2));
  float*    pdec = (float*)   (ws + alloc((size_t)B_ * A_ * 4));
  float*    xbuf = (float*)   (ws + alloc((size_t)B_ * X_ * 4));
  _Float16* x16  = (_Float16*)(ws + alloc((size_t)B_ * X_ * 2));
  float*    gi   = (float*)   (ws + alloc((size_t)B_ * G_ * 4));
  float*    gh   = (float*)   (ws + alloc((size_t)B_ * G_ * 4));

  auto cvt = [&](const float* s, _Float16* d, int n) {
    cvt_f32_to_f16<<<(n + 255) / 256, 256, 0, stream>>>(s, d, n);
  };
  cvt(W_enc, W_enc_h, A_ * E_);
  cvt(W_dec, W_dec_h, A_ * H_);
  cvt(W_ih,  W_ih_h,  G_ * X_);
  cvt(W_hh,  W_hh_h,  G_ * H_);
  cvt(W_out, W_out_h, V_ * H_);

  zero_f32<<<(B_ * H_) / 256, 256, 0, stream>>>(h, B_ * H_);
  zero_f16<<<(B_ * H_) / 256, 256, 0, stream>>>(h16, B_ * H_);
  gather_embs<<<(B_ * L_ * EMB_) / 256, 256, 0, stream>>>(targets, emb_table, embs);

  // proj_enc = enc_out @ W_enc^T : M=B*T=32768, N=256, K=512 (one-time)
  gemm_wmma<<<dim3(A_ / 64, (B_ * T_) / 16), 128, 0, stream>>>(
      enc_out, E_, W_enc_h, E_, proj_enc, A_, E_);

  for (int l = 0; l < L_; ++l) {
    // proj_dec = h @ W_dec^T : (64x256, K=512)
    gemm_skinny<<<A_ / 64, 128, 0, stream>>>(h16, H_, W_dec_h, H_, nullptr,
                                             pdec, A_, H_);
    // energy -> softmax -> context -> x ; writes attn output
    attn_fused<<<B_, 256, 0, stream>>>(proj_enc, pdec, v, enc_out, embs, xbuf,
                                       out_attns, l);
    // x -> f16 for the gi GEMM
    cvt_f32_to_f16<<<(B_ * X_) / 256, 256, 0, stream>>>(xbuf, x16, B_ * X_);
    // gi = x @ W_ih^T + b_ih : (64x1536, K=576)
    gemm_skinny<<<G_ / 64, 128, 0, stream>>>(x16, X_, W_ih_h, X_, b_ih,
                                             gi, G_, X_);
    // gh = h @ W_hh^T + b_hh : (64x1536, K=512)
    gemm_skinny<<<G_ / 64, 128, 0, stream>>>(h16, H_, W_hh_h, H_, b_hh,
                                             gh, G_, H_);
    // GRU gate math -> h (f32 + f16 copies)
    gru_update<<<(B_ * H_) / 256, 256, 0, stream>>>(gi, gh, h, h16);
    // logits = h @ W_out^T + b_out : (64x8000, K=512); C strided into (B,L,V)
    gemm_skinny<<<V_ / 64, 128, 0, stream>>>(h16, H_, W_out_h, H_, b_out,
                                             out_logits + (size_t)l * V_,
                                             (long long)L_ * V_, H_);
  }
}
